// HSTULayer_4587025072279
// MI455X (gfx1250) — matrix operand order repaired and checked by reading, not compiled, not run on hardware
//
#include <hip/hip_runtime.h>
#include <hip/hip_bf16.h>

typedef __attribute__((ext_vector_type(16))) _Float16 v16h;
typedef __attribute__((ext_vector_type(8)))  _Float16 v8h;
typedef __attribute__((ext_vector_type(8)))  float    v8f;
typedef int v4i_ __attribute__((vector_size(16)));   // clang generic vector: matches builtin's V4i

union Frag16 { v16h v; v8h h[2]; };

#define WMMA_F16(a, b, c) \
    __builtin_amdgcn_wmma_f32_16x16x32_f16(false, (a), false, (b), (short)0, (c), false, false)

#ifndef __has_builtin
#define __has_builtin(x) 0
#endif

#if defined(__HIP_DEVICE_COMPILE__) && __has_builtin(__builtin_amdgcn_global_load_async_to_lds_b128)
#define HAVE_ASYNC_LDS 1
#else
#define HAVE_ASYNC_LDS 0
#endif

// Contiguous 16B global -> LDS copy; async (ASYNCcnt, no VGPR round-trip) when
// the gfx1250 builtin exists, else a plain b128 copy.
__device__ __forceinline__ void ldg_b128_to_lds(_Float16* lds_dst, const _Float16* gsrc)
{
#if HAVE_ASYNC_LDS
    __builtin_amdgcn_global_load_async_to_lds_b128(
        (__attribute__((address_space(1))) v4i_*)gsrc,
        (__attribute__((address_space(3))) v4i_*)lds_dst, 0, 0);
#else
    *(float4*)lds_dst = *(const float4*)gsrc;
#endif
}

__device__ __forceinline__ void async_wait_all()
{
#if HAVE_ASYNC_LDS
#if __has_builtin(__builtin_amdgcn_s_wait_asynccnt)
    __builtin_amdgcn_s_wait_asynccnt(0);
#else
    asm volatile("s_wait_asynccnt 0x0" ::: "memory");
#endif
#endif
}

// ---------------------------------------------------------------------------
// Problem constants
// ---------------------------------------------------------------------------
static constexpr int Bsz = 2;
static constexpr int S   = 2048;
static constexpr int D   = 1024;
static constexpr int H   = 16;
static constexpr int HD  = 64;
static constexpr int M_TOT = Bsz * S;          // 4096 rows

// ---------------------------------------------------------------------------
// Kernel 1: LayerNorm (f32 in) -> f16 xn.  One block per row.
// ---------------------------------------------------------------------------
__global__ __launch_bounds__(256)
void ln_kernel(const float* __restrict__ x, const float* __restrict__ g,
               const float* __restrict__ be, _Float16* __restrict__ o)
{
    __shared__ float rs[256];
    __shared__ float rq[256];
    const int row = blockIdx.x;
    const int tid = threadIdx.x;
    const float* xr = x + (size_t)row * D;

    float4 v = *(const float4*)(xr + tid * 4);
    rs[tid] = v.x + v.y + v.z + v.w;
    rq[tid] = v.x * v.x + v.y * v.y + v.z * v.z + v.w * v.w;
    __syncthreads();
    for (int st = 128; st > 0; st >>= 1) {
        if (tid < st) { rs[tid] += rs[tid + st]; rq[tid] += rq[tid + st]; }
        __syncthreads();
    }
    const float mu   = rs[0] * (1.0f / (float)D);
    const float var  = rq[0] * (1.0f / (float)D) - mu * mu;
    const float rstd = rsqrtf(var + 1e-5f);

    const float4 gv = *(const float4*)(g  + tid * 4);
    const float4 bv = *(const float4*)(be + tid * 4);
    union { _Float16 h[4]; float2 f; } ou;
    ou.h[0] = (_Float16)((v.x - mu) * rstd * gv.x + bv.x);
    ou.h[1] = (_Float16)((v.y - mu) * rstd * gv.y + bv.y);
    ou.h[2] = (_Float16)((v.z - mu) * rstd * gv.z + bv.z);
    ou.h[3] = (_Float16)((v.w - mu) * rstd * gv.w + bv.w);
    *(float2*)(o + (size_t)row * D + tid * 4) = ou.f;
}

// ---------------------------------------------------------------------------
// Kernel 2: tiled transpose + f32->f16:  in[K][N] -> out[N][K].
// 32x32 tiles through LDS (padded to kill bank conflicts).
// ---------------------------------------------------------------------------
__global__ __launch_bounds__(256)
void cvt_t_kernel(const float* __restrict__ a, _Float16* __restrict__ o,
                  int K, int N)
{
    __shared__ float tile[32][33];
    const int k0 = blockIdx.y * 32, n0 = blockIdx.x * 32;
    const int tx = threadIdx.x & 31, ty = threadIdx.x >> 5;   // 32x8
#pragma unroll
    for (int i = 0; i < 32; i += 8)
        tile[ty + i][tx] = a[(size_t)(k0 + ty + i) * N + n0 + tx];
    __syncthreads();
#pragma unroll
    for (int i = 0; i < 32; i += 8)
        o[(size_t)(n0 + ty + i) * K + k0 + tx] = (_Float16)tile[tx][ty + i];
}

// ---------------------------------------------------------------------------
// Kernel 3: WMMA GEMM  C[M,N] = A[M,K] * W[K,N], with W given PRE-TRANSPOSED
// as Wt[N][K] (f16).  Block tile 64(M) x 128(N), 8 waves (2x4), each wave a
// 2x2 grid of 16x16 WMMA tiles (A/B fragment reuse).  K-step 32.
// All LDS staging is contiguous b128 (async when available).
//
// MODE 0: +bias, scatter f16 into q/k [B,H,S,HD] and v TRANSPOSED [B,H,HD,S]
// MODE 1: +bias, SiLU, store f32 (gate path u)
// MODE 2: +bias + residual x, store f32 (final output)
// ---------------------------------------------------------------------------
template<int MODE>
__global__ __launch_bounds__(256)
void gemm16(const _Float16* __restrict__ A, const _Float16* __restrict__ Wt,
            const float* __restrict__ bias, int M, int N, int K,
            _Float16* __restrict__ qh, _Float16* __restrict__ kh,
            _Float16* __restrict__ vh,
            float* __restrict__ u_out,
            const float* __restrict__ xres, float* __restrict__ out)
{
    __shared__ __align__(16) _Float16 As[64 * 32];    // [m][k]
    __shared__ __align__(16) _Float16 Bs[128 * 32];   // [n][k]

    const int tid  = threadIdx.x;
    const int w    = tid >> 5;
    const int lane = tid & 31;
    const int l15  = lane & 15;
    const int hi2  = lane >> 4;
    const int wm2  = w & 1;        // 0..1  -> 32-row slab
    const int wn2  = w >> 1;       // 0..3  -> 32-col slab
    const int bm   = blockIdx.y * 64;
    const int bn   = blockIdx.x * 128;

    // A loader: 256 chunks of 16B  (64 rows x 32 halves)
    const int amr = tid >> 2, amc = (tid & 3) * 8;
    v8f acc[2][2];
#pragma unroll
    for (int i = 0; i < 2; i++)
#pragma unroll
        for (int j = 0; j < 2; j++) acc[i][j] = (v8f){};

    for (int k0 = 0; k0 < K; k0 += 32) {
        ldg_b128_to_lds(&As[amr * 32 + amc],
                        A + (size_t)(bm + amr) * K + k0 + amc);
        // B loader: 512 chunks of 16B (128 rows x 32 halves), 2 per thread
#pragma unroll
        for (int c = 0; c < 2; c++) {
            const int j = tid * 2 + c;
            const int br = j >> 2, bc = (j & 3) * 8;
            ldg_b128_to_lds(&Bs[br * 32 + bc],
                            Wt + (size_t)(bn + br) * K + k0 + bc);
        }
        if (k0 + 32 < K) {
            __builtin_prefetch(A  + (size_t)(bm + amr) * K + k0 + 32 + amc, 0, 1);
            __builtin_prefetch(Wt + (size_t)(bn + tid / 2) * K + k0 + 32, 0, 1);
        }
        async_wait_all();
        __syncthreads();

        Frag16 af[2], bf[2];
        const int ka = hi2 ? 8 : 0;
        const int kb = hi2 ? 8 : 0;
#pragma unroll
        for (int mt = 0; mt < 2; mt++) {
            const int arow = wm2 * 32 + mt * 16 + l15;
            af[mt].h[0] = *(const v8h*)&As[arow * 32 + ka];
            af[mt].h[1] = *(const v8h*)&As[arow * 32 + ka + 16];
        }
#pragma unroll
        for (int nt = 0; nt < 2; nt++) {
            const int brow = wn2 * 32 + nt * 16 + l15;
            bf[nt].h[0] = *(const v8h*)&Bs[brow * 32 + kb];
            bf[nt].h[1] = *(const v8h*)&Bs[brow * 32 + kb + 16];
        }
#pragma unroll
        for (int mt = 0; mt < 2; mt++)
#pragma unroll
            for (int nt = 0; nt < 2; nt++)
                acc[mt][nt] = WMMA_F16(af[mt].v, bf[nt].v, acc[mt][nt]);
        __syncthreads();
    }

    // ------------------------- epilogue -------------------------
#pragma unroll
    for (int nt = 0; nt < 2; nt++) {
        const int gn = bn + wn2 * 32 + nt * 16 + l15;
        const float bv = bias[gn];
#pragma unroll
        for (int mt = 0; mt < 2; mt++) {
#pragma unroll
            for (int r = 0; r < 8; r++) {
                const int gm = bm + wm2 * 32 + mt * 16 + r + 8 * hi2;
                float val = acc[mt][nt][r] + bv;
                if (MODE == 0) {
                    const int part = gn >> 10;          // 0=q 1=k 2=v
                    const int rem  = gn & 1023;
                    const int hh   = rem >> 6;
                    const int hd   = rem & 63;
                    const int bb   = gm >> 11;          // batch
                    const int ss   = gm & 2047;         // seq
                    if (part == 0) {
                        qh[(((size_t)(bb * H + hh)) * S + ss) * HD + hd] = (_Float16)val;
                    } else if (part == 1) {
                        kh[(((size_t)(bb * H + hh)) * S + ss) * HD + hd] = (_Float16)val;
                    } else {  // v stored transposed: [B,H,HD,S]
                        vh[(((size_t)(bb * H + hh)) * HD + hd) * S + ss] = (_Float16)val;
                    }
                } else if (MODE == 1) {
                    u_out[(size_t)gm * N + gn] = val / (1.0f + __expf(-val));
                } else {
                    out[(size_t)gm * N + gn] = val + xres[(size_t)gm * N + gn];
                }
            }
        }
    }
}

// ---------------------------------------------------------------------------
// Kernel 4: sigmoid attention fused with the SiLU gate multiply.
// Grid: (S/128, B*H).  8 waves; wave w owns q rows [qb*128 + w*16, +16).
// V comes in pre-transposed [B,H,HD,S], so every LDS tile is a straight copy.
// ---------------------------------------------------------------------------
__global__ __launch_bounds__(256)
void attn_kernel(const _Float16* __restrict__ qh, const _Float16* __restrict__ kh,
                 const _Float16* __restrict__ vh, const float* __restrict__ u,
                 const float* __restrict__ relb, _Float16* __restrict__ ctxu)
{
    __shared__ __align__(16) _Float16 qs[128 * 64];      // q block [row][hd]
    __shared__ __align__(16) _Float16 ks[32 * 64];       // k tile  [kv][hd]
    __shared__ __align__(16) _Float16 vt[64 * 32];       // v tile  [hd][kv]
    __shared__ __align__(16) _Float16 ps[8 * 16 * 32];   // per-wave prob tile

    const int bh  = blockIdx.y;                // b*H + h
    const int bat = bh >> 4;
    const int hh  = bh & 15;
    const int qb  = blockIdx.x;                // 0..15 (128-row q blocks)
    const int tid = threadIdx.x;
    const int w    = tid >> 5;
    const int lane = tid & 31;
    const int l15  = lane & 15;
    const int hi2  = lane >> 4;

    const _Float16* qbase = qh + (size_t)bh * S * HD;    // [s][hd]
    const _Float16* kbase = kh + (size_t)bh * S * HD;    // [s][hd]
    const _Float16* vbase = vh + (size_t)bh * HD * S;    // [hd][s]  (transposed)

    // stage 128x64 q block: 1024 chunks of 16B, 4 per thread
#pragma unroll
    for (int c = 0; c < 4; c++) {
        const int j = c * 256 + tid;
        const int r = j >> 3, col = (j & 7) * 8;
        ldg_b128_to_lds(&qs[r * 64 + col],
                        qbase + (size_t)(qb * 128 + r) * HD + col);
    }
    async_wait_all();
    __syncthreads();

    const float bias_h = relb[hh];
    const float scale  = 0.125f;               // HD^-0.5

    v8f acco[4];
#pragma unroll
    for (int t = 0; t < 4; t++) acco[t] = (v8f){};

    const int q_lo = qb * 128 + w * 16;
    const int nkv  = (qb + 1) * 4;             // causal: kv tiles of 32

    for (int kvb = 0; kvb < nkv; kvb++) {
        const int kv0 = kvb * 32;
        // k tile [kv][hd]: 256 chunks of 16B
        ldg_b128_to_lds(&ks[(tid >> 3) * 64 + (tid & 7) * 8],
                        kbase + (size_t)(kv0 + (tid >> 3)) * HD + (tid & 7) * 8);
        // v tile [hd][kv]: straight rows of the transposed V
        ldg_b128_to_lds(&vt[(tid >> 2) * 32 + (tid & 3) * 8],
                        vbase + (size_t)(tid >> 2) * S + kv0 + (tid & 3) * 8);
        async_wait_all();
        __syncthreads();

        // -------- scores = q * k^T  (16 q rows x 32 kv cols) --------
        v8f accs[2];
        accs[0] = (v8f){}; accs[1] = (v8f){};
#pragma unroll
        for (int nt = 0; nt < 2; nt++) {
#pragma unroll
            for (int kstep = 0; kstep < 2; kstep++) {
                Frag16 a, b;
                const int arow = w * 16 + l15;
                const int ka   = (hi2 ? 8 : 0) + kstep * 32;
                a.h[0] = *(const v8h*)&qs[arow * 64 + ka];
                a.h[1] = *(const v8h*)&qs[arow * 64 + ka + 16];
                const int brow = nt * 16 + l15;
                const int kb   = (hi2 ? 8 : 0) + kstep * 32;
                b.h[0] = *(const v8h*)&ks[brow * 64 + kb];
                b.h[1] = *(const v8h*)&ks[brow * 64 + kb + 16];
                accs[nt] = WMMA_F16(a.v, b.v, accs[nt]);
            }
        }

        // -------- scale + head bias + sigmoid + causal mask; transpose to
        //          A-fragment layout via per-wave LDS scratch --------
#pragma unroll
        for (int nt = 0; nt < 2; nt++) {
#pragma unroll
            for (int r = 0; r < 8; r++) {
                const int qi = q_lo + r + 8 * hi2;
                const int kj = kv0 + nt * 16 + l15;
                const float sc = accs[nt][r] * scale + bias_h;
                const float p  = (kj <= qi) ? (1.0f / (1.0f + __expf(-sc))) : 0.0f;
                ps[w * 512 + (r + 8 * hi2) * 32 + nt * 16 + l15] = (_Float16)p;
            }
        }
        Frag16 pa;
        const int kp = hi2 ? 8 : 0;
        pa.h[0] = *(const v8h*)&ps[w * 512 + l15 * 32 + kp];
        pa.h[1] = *(const v8h*)&ps[w * 512 + l15 * 32 + kp + 16];

        // -------- ctx += attn * v  (16 x 64, four 16x16 N-tiles) --------
#pragma unroll
        for (int ht = 0; ht < 4; ht++) {
            Frag16 vb;
            const int vrow = ht * 16 + l15;
            const int kvh  = hi2 ? 8 : 0;
            vb.h[0] = *(const v8h*)&vt[vrow * 32 + kvh];
            vb.h[1] = *(const v8h*)&vt[vrow * 32 + kvh + 16];
            acco[ht] = WMMA_F16(pa.v, vb.v, acco[ht]);
        }
        __syncthreads();   // before next kv tile overwrites ks/vt
    }

    // -------- fuse gate: ctxu = ctx * u, store f16 for the out GEMM ------
#pragma unroll
    for (int ht = 0; ht < 4; ht++) {
#pragma unroll
        for (int r = 0; r < 8; r++) {
            const int ss  = q_lo + r + 8 * hi2;
            const int col = hh * HD + ht * 16 + l15;
            const size_t idx = ((size_t)(bat * S + ss)) * D + col;
            ctxu[idx] = (_Float16)(acco[ht][r] * u[idx]);
        }
    }
}

// ---------------------------------------------------------------------------
// Host launcher
// ---------------------------------------------------------------------------
extern "C" void kernel_launch(void* const* d_in, const int* in_sizes, int n_in,
                              void* d_out, int out_size, void* d_ws, size_t ws_size,
                              hipStream_t stream)
{
    (void)in_sizes; (void)n_in; (void)out_size; (void)ws_size;
    const float* x      = (const float*)d_in[0];
    // d_in[1] attention_mask: causal tril — computed analytically in-kernel
    const float* gamma  = (const float*)d_in[2];
    const float* beta   = (const float*)d_in[3];
    const float* w_qkv  = (const float*)d_in[4];
    const float* b_qkv  = (const float*)d_in[5];
    const float* w_gate = (const float*)d_in[6];
    const float* b_gate = (const float*)d_in[7];
    const float* w_out  = (const float*)d_in[8];
    const float* b_out  = (const float*)d_in[9];
    const float* relb   = (const float*)d_in[10];

    char* ws = (char*)d_ws;
    size_t off = 0;
    auto carve = [&](size_t bytes) {
        void* p = ws + off;
        off += (bytes + 255) & ~(size_t)255;
        return p;
    };
    _Float16* xn_h   = (_Float16*)carve((size_t)M_TOT * D * 2);       // 8 MB
    _Float16* wqkvT  = (_Float16*)carve((size_t)D * 3 * D * 2);       // 6 MB  [3D][D]
    _Float16* wgateT = (_Float16*)carve((size_t)D * D * 2);           // 2 MB  [D][D]
    _Float16* woutT  = (_Float16*)carve((size_t)D * D * 2);           // 2 MB  [D][D]
    _Float16* qh     = (_Float16*)carve((size_t)M_TOT * D * 2);       // 8 MB
    _Float16* kh     = (_Float16*)carve((size_t)M_TOT * D * 2);       // 8 MB
    _Float16* vh     = (_Float16*)carve((size_t)M_TOT * D * 2);       // 8 MB (transposed)
    float*    u_f    = (float*)   carve((size_t)M_TOT * D * 4);       // 16 MB
    _Float16* ctxu_h = (_Float16*)carve((size_t)M_TOT * D * 2);       // 8 MB

    // 1) LayerNorm -> f16
    ln_kernel<<<M_TOT, 256, 0, stream>>>(x, gamma, beta, xn_h);

    // 2) weight transpose + down-conversion:  W[K][N] -> Wt[N][K] f16
    cvt_t_kernel<<<dim3(3 * D / 32, D / 32), 256, 0, stream>>>(w_qkv,  wqkvT,  D, 3 * D);
    cvt_t_kernel<<<dim3(D / 32, D / 32),     256, 0, stream>>>(w_gate, wgateT, D, D);
    cvt_t_kernel<<<dim3(D / 32, D / 32),     256, 0, stream>>>(w_out,  woutT,  D, D);

    // 3) QKV projection (scatter per-head; v transposed)
    gemm16<0><<<dim3(3 * D / 128, M_TOT / 64), 256, 0, stream>>>(
        xn_h, wqkvT, b_qkv, M_TOT, 3 * D, D,
        qh, kh, vh, nullptr, nullptr, nullptr);

    // 4) SiLU gate:  u = silu(xn @ w_gate + b_gate)
    gemm16<1><<<dim3(D / 128, M_TOT / 64), 256, 0, stream>>>(
        xn_h, wgateT, b_gate, M_TOT, D, D,
        nullptr, nullptr, nullptr, u_f, nullptr, nullptr);

    // 5) sigmoid attention + gate fusion -> ctxu (f16)
    attn_kernel<<<dim3(S / 128, Bsz * H), 256, 0, stream>>>(
        qh, kh, vh, u_f, relb, ctxu_h);

    // 6) output projection + bias + residual -> d_out (f32)
    gemm16<2><<<dim3(D / 128, M_TOT / 64), 256, 0, stream>>>(
        ctxu_h, woutT, b_out, M_TOT, D, D,
        nullptr, nullptr, nullptr, nullptr, x, (float*)d_out);
}
